// MultiHeadAttention_21423296873196
// MI455X (gfx1250) — matrix operand order repaired and checked by reading, not compiled, not run on hardware
//
#include <hip/hip_runtime.h>
#include <hip/hip_bf16.h>

// ---------------------------------------------------------------------------
// Types
// ---------------------------------------------------------------------------
typedef __attribute__((ext_vector_type(16))) __bf16        v16bf;
typedef __attribute__((ext_vector_type(8)))  float         v8f;
typedef __attribute__((ext_vector_type(4)))  unsigned int  u32x4;
typedef __attribute__((ext_vector_type(2)))  unsigned int  u32x2;
typedef __attribute__((ext_vector_type(4)))  float         f32x4;

// 16 bf16 lane-fragment for WMMA A/B operands (two 16B chunks)
union FragA {
    u32x4 q[2];
    v16bf v;
};

__device__ __forceinline__ unsigned short f32_to_bf16(float f) {
    unsigned int u = __float_as_uint(f);
    unsigned int r = u + 0x7FFFu + ((u >> 16) & 1u);  // round-to-nearest-even
    return (unsigned short)(r >> 16);
}

__device__ __forceinline__ v8f wmma_bf16(const FragA& a, const FragA& b, v8f c) {
    return __builtin_amdgcn_wmma_f32_16x16x32_bf16(
        /*neg_a=*/false, a.v, /*neg_b=*/false, b.v,
        /*c_mod=*/(short)0, c, /*reuse_a=*/false, /*reuse_b=*/false);
}

// Async global->LDS copy of 16B per lane (CDNA5 GLOBAL_LOAD_ASYNC_TO_LDS_B128,
// GV addressing: 64-bit vaddr pair, saddr=off). Tracked by ASYNCcnt.
__device__ __forceinline__ void async_ld_b128(unsigned int lds_off, const void* gptr) {
    asm volatile("global_load_async_to_lds_b128 %0, %1, off"
                 :
                 : "v"(lds_off), "v"((unsigned long long)(size_t)gptr)
                 : "memory");
}

__device__ __forceinline__ void wait_asynccnt0() {
    asm volatile("s_wait_asynccnt 0" ::: "memory");
}

// ---------------------------------------------------------------------------
// Problem constants
// ---------------------------------------------------------------------------
#define MB_B 2
#define MB_S 2048
#define MB_D 1024
#define MB_H 16
#define MB_DH 64

// ---------------------------------------------------------------------------
// Kernel 1: fp32 -> bf16 convert (4 elems / thread)
// ---------------------------------------------------------------------------
__global__ __launch_bounds__(256) void cvt_f32_bf16x4(const float* __restrict__ x,
                                                      unsigned short* __restrict__ y,
                                                      int n4) {
    int i = blockIdx.x * blockDim.x + threadIdx.x;
    if (i >= n4) return;
    f32x4 f = *(const f32x4*)(x + 4 * (size_t)i);
    union { unsigned short u[4]; u32x2 v; } o;
    o.u[0] = f32_to_bf16(f.x);
    o.u[1] = f32_to_bf16(f.y);
    o.u[2] = f32_to_bf16(f.z);
    o.u[3] = f32_to_bf16(f.w);
    *(u32x2*)(y + 4 * (size_t)i) = o.v;
}

// ---------------------------------------------------------------------------
// Kernel 2: NT GEMM  C[M,N] = A[M,K] * B[N,K]^T, both bf16 row-major.
// Tile 128x128, BK=32. 256 threads = 8 waves in 2(M) x 4(N); each wave 64x32.
// Staging uses async global->LDS b128 copies (no VGPR round trip).
// ---------------------------------------------------------------------------
#define BM 128
#define BN 128
#define BK 32
#define LDT 40   // LDS row stride in ushorts (80B, 16B-aligned rows)

__global__ __launch_bounds__(256) void gemm_bf16nt(const unsigned short* __restrict__ A,
                                                   const unsigned short* __restrict__ Bw,
                                                   float* __restrict__ C,
                                                   int M, int N, int K) {
    __shared__ __align__(16) unsigned short sA[BM * LDT];
    __shared__ __align__(16) unsigned short sB[BN * LDT];

    const int tid  = threadIdx.x;
    const int lane = tid & 31;
    const int w    = tid >> 5;
    const int wm   = w >> 2;      // 0..1
    const int wn   = w & 3;       // 0..3
    const int g    = lane >> 4;   // lane half
    const int hn   = lane & 15;

    const int tileM = blockIdx.y * BM;
    const int tileN = blockIdx.x * BN;

    v8f acc[4][2] = {};

    for (int k0 = 0; k0 < K; k0 += BK) {
        // ---- stage A,B tiles into LDS: 512 x 16B chunks each, 2/thread ----
#pragma unroll
        for (int i = 0; i < 2; ++i) {
            int c = tid + 256 * i;            // chunk id 0..511
            int r = c >> 2;                   // tile row 0..127
            int q = (c & 3) * 8;              // ushort offset within 32-col row
            async_ld_b128((unsigned int)(size_t)&sA[r * LDT + q],
                          &A[(size_t)(tileM + r) * K + k0 + q]);
            async_ld_b128((unsigned int)(size_t)&sB[r * LDT + q],
                          &Bw[(size_t)(tileN + r) * K + k0 + q]);
        }
        // prefetch next K tile into L2 while this one is consumed
        if (k0 + BK < K) {
            __builtin_prefetch(&A[(size_t)(tileM + (tid >> 1)) * K + k0 + BK], 0, 0);
            __builtin_prefetch(&Bw[(size_t)(tileN + (tid >> 1)) * K + k0 + BK], 0, 0);
        }
        wait_asynccnt0();     // our chunks are in LDS
        __syncthreads();      // everyone's chunks are in LDS

        // ---- fragments ----
        FragA af[4], bf[2];
#pragma unroll
        for (int mi = 0; mi < 4; ++mi) {
            int row = wm * 64 + mi * 16 + hn;     // A lane row (M)
            af[mi].q[0] = *(const u32x4*)&sA[row * LDT + 8 * g];
            af[mi].q[1] = *(const u32x4*)&sA[row * LDT + 16 + 8 * g];
        }
#pragma unroll
        for (int ni = 0; ni < 2; ++ni) {
            int col = wn * 32 + ni * 16 + hn;     // B lane col (N)
            bf[ni].q[0] = *(const u32x4*)&sB[col * LDT + 8 * g];
            bf[ni].q[1] = *(const u32x4*)&sB[col * LDT + 16 + 8 * g];
        }
#pragma unroll
        for (int mi = 0; mi < 4; ++mi)
#pragma unroll
            for (int ni = 0; ni < 2; ++ni)
                acc[mi][ni] = wmma_bf16(af[mi], bf[ni], acc[mi][ni]);
        __syncthreads();
    }

    // ---- epilogue: C layout lane = col hn, rows r + 8g ----
#pragma unroll
    for (int mi = 0; mi < 4; ++mi)
#pragma unroll
        for (int ni = 0; ni < 2; ++ni) {
            size_t base = (size_t)(tileM + wm * 64 + mi * 16 + 8 * g) * N
                        + (tileN + wn * 32 + ni * 16 + hn);
#pragma unroll
            for (int r = 0; r < 8; ++r)
                C[base + (size_t)r * N] = acc[mi][ni][r];
        }
}

// ---------------------------------------------------------------------------
// Kernel 3: RoPE on q,k + pack to WMMA-friendly bf16 layouts.
//   Qb, Kb : [b][h][s][dh]   (roped)
//   Vt     : [b][h][dh][s]   (transposed so PV B-fragments are contiguous)
// ---------------------------------------------------------------------------
__global__ __launch_bounds__(256) void rope_pack(const float* __restrict__ qkv,
                                                 unsigned short* __restrict__ Qb,
                                                 unsigned short* __restrict__ Kb,
                                                 unsigned short* __restrict__ Vt) {
    int idx = blockIdx.x * blockDim.x + threadIdx.x;  // B*H*S*dh threads
    int d = idx & (MB_DH - 1);
    int t = idx >> 6;
    int s = t % MB_S;
    int hb = t / MB_S;
    int h = hb % MB_H;
    int b = hb / MB_H;

    size_t m = (size_t)b * MB_S + s;
    const float* qrow = qkv + m * (3 * MB_D);

    int j = d >> 1;
    float inv = __powf(10000.0f, -(float)(2 * j) / (float)MB_DH);
    float ang = (float)s * inv;
    float cs = __cosf(ang);
    float sn = __sinf(ang);

    int fe = h * MB_DH + (j << 1);
    float qe = qrow[fe], qo = qrow[fe + 1];
    float ke = qrow[MB_D + fe], ko = qrow[MB_D + fe + 1];
    float qv = (d & 1) ? (qe * sn + qo * cs) : (qe * cs - qo * sn);
    float kv = (d & 1) ? (ke * sn + ko * cs) : (ke * cs - ko * sn);
    float vv = qrow[2 * MB_D + h * MB_DH + d];

    size_t bhs = ((size_t)(b * MB_H + h) * MB_S + s) * MB_DH + d;
    Qb[bhs] = f32_to_bf16(qv);
    Kb[bhs] = f32_to_bf16(kv);
    Vt[((size_t)(b * MB_H + h) * MB_DH + d) * MB_S + s] = f32_to_bf16(vv);
}

// ---------------------------------------------------------------------------
// Kernel 4: causal flash attention, one wave per 16-query block.
//   4 waves / WG, no cross-wave sync (trip counts differ under causality).
//   Scores: 4x wmma per 32-key block; softmax in fp32 with width-16 shfl
//   reductions; P transposed through per-wave LDS (s_wait_dscnt 0 guards the
//   intra-wave LDS RAW); PV: 4x wmma into 16x64 f32 accumulators.
// ---------------------------------------------------------------------------
__global__ __launch_bounds__(128) void flash_attn(const unsigned short* __restrict__ Qb,
                                                  const unsigned short* __restrict__ Kb,
                                                  const unsigned short* __restrict__ Vt,
                                                  unsigned short* __restrict__ Obf) {
    __shared__ __align__(16) unsigned short pT[4][16 * 32];

    const int lane = threadIdx.x & 31;
    const int w    = threadIdx.x >> 5;
    const int g    = lane >> 4;
    const int hn   = lane & 15;

    const int wgs_per_bh = MB_S / 64;                 // 32 WGs per (b,h)
    const int bh     = blockIdx.x / wgs_per_bh;
    const int qchunk = blockIdx.x % wgs_per_bh;
    const int qbase  = (qchunk * 4 + w) * 16;
    const int b = bh / MB_H;
    const int h = bh % MB_H;

    const unsigned short* Qbase = Qb + ((size_t)bh * MB_S + qbase) * MB_DH;
    const unsigned short* Kbase = Kb + (size_t)bh * MB_S * MB_DH;
    const unsigned short* Vbase = Vt + (size_t)bh * MB_DH * MB_S;

    // Q fragments for the two dh k-steps (hoisted; reused every key block)
    FragA qf[2];
#pragma unroll
    for (int kk = 0; kk < 2; ++kk) {
        const unsigned short* p = Qbase + hn * MB_DH + kk * 32 + 8 * g;
        qf[kk].q[0] = *(const u32x4*)p;
        qf[kk].q[1] = *(const u32x4*)(p + 16);
    }

    v8f   Oa[4] = {};
    float mrun[8], lrun[8];
#pragma unroll
    for (int r = 0; r < 8; ++r) { mrun[r] = -1e30f; lrun[r] = 0.0f; }

    unsigned short* myP = pT[w];
    const float scale = 0.125f;   // 1/sqrt(64)
    const int nkb = (qbase + 15) / 32 + 1;

    for (int kb = 0; kb < nkb; ++kb) {
        const int key0 = kb * 32;

        // ---- scores S = Q K^T for 32 keys: two 16x16 tiles ----
        v8f s0 = {}, s1 = {};
#pragma unroll
        for (int kk = 0; kk < 2; ++kk) {
            FragA kf0, kf1;
            const unsigned short* p0 =
                Kbase + (size_t)(key0 + hn) * MB_DH + kk * 32 + 8 * g;
            kf0.q[0] = *(const u32x4*)p0;
            kf0.q[1] = *(const u32x4*)(p0 + 16);
            const unsigned short* p1 =
                Kbase + (size_t)(key0 + 16 + hn) * MB_DH + kk * 32 + 8 * g;
            kf1.q[0] = *(const u32x4*)p1;
            kf1.q[1] = *(const u32x4*)(p1 + 16);
            s0 = wmma_bf16(qf[kk], kf0, s0);
            s1 = wmma_bf16(qf[kk], kf1, s1);
        }

        // ---- online softmax (row-wise across the 16-lane half) ----
#pragma unroll
        for (int r = 0; r < 8; ++r) {
            int row = qbase + 8 * g + r;
            int c0 = key0 + hn;
            int c1 = key0 + 16 + hn;
            float a0 = (c0 <= row) ? s0[r] * scale : -1e30f;
            float a1 = (c1 <= row) ? s1[r] * scale : -1e30f;
            float mx = fmaxf(a0, a1);
#pragma unroll
            for (int off = 8; off >= 1; off >>= 1)
                mx = fmaxf(mx, __shfl_xor(mx, off, 16));
            float mnew  = fmaxf(mrun[r], mx);
            float alpha = __expf(mrun[r] - mnew);
            float p0 = __expf(a0 - mnew);
            float p1 = __expf(a1 - mnew);
            float rs = p0 + p1;
#pragma unroll
            for (int off = 8; off >= 1; off >>= 1)
                rs += __shfl_xor(rs, off, 16);
            lrun[r] = lrun[r] * alpha + rs;
            mrun[r] = mnew;
            Oa[0][r] *= alpha; Oa[1][r] *= alpha;
            Oa[2][r] *= alpha; Oa[3][r] *= alpha;
            int trow = 8 * g + r;
            myP[trow * 32 + hn]      = f32_to_bf16(p0);
            myP[trow * 32 + 16 + hn] = f32_to_bf16(p1);
        }

        // intra-wave LDS RAW: wait for the transposed P stores
        asm volatile("s_wait_dscnt 0" ::: "memory");

        // ---- P as A-fragment (K = 32 keys) ----
        FragA pf;
        pf.q[0] = *(const u32x4*)&myP[hn * 32 + 8 * g];
        pf.q[1] = *(const u32x4*)&myP[hn * 32 + 16 + 8 * g];

        // ---- O += P V : 4 dh tiles ----
#pragma unroll
        for (int nt = 0; nt < 4; ++nt) {
            FragA vf;
            const unsigned short* vp =
                Vbase + (size_t)(16 * nt + hn) * MB_S + key0 + 8 * g;
            vf.q[0] = *(const u32x4*)vp;
            vf.q[1] = *(const u32x4*)(vp + 16);
            Oa[nt] = wmma_bf16(pf, vf, Oa[nt]);
        }
    }

    // ---- normalize and write bf16 [b][s][h*dh + col] ----
#pragma unroll
    for (int r = 0; r < 8; ++r) {
        float inv = 1.0f / lrun[r];
        int row = qbase + 8 * g + r;
        size_t base = ((size_t)b * MB_S + row) * MB_D + h * MB_DH + hn;
        Obf[base + 0]  = f32_to_bf16(Oa[0][r] * inv);
        Obf[base + 16] = f32_to_bf16(Oa[1][r] * inv);
        Obf[base + 32] = f32_to_bf16(Oa[2][r] * inv);
        Obf[base + 48] = f32_to_bf16(Oa[3][r] * inv);
    }
}

// ---------------------------------------------------------------------------
// Launch
// ---------------------------------------------------------------------------
extern "C" void kernel_launch(void* const* d_in, const int* in_sizes, int n_in,
                              void* d_out, int out_size, void* d_ws, size_t ws_size,
                              hipStream_t stream) {
    (void)in_sizes; (void)n_in; (void)out_size; (void)ws_size;
    const float* X    = (const float*)d_in[0];   // [B,S,D]
    const float* Wqkv = (const float*)d_in[1];   // [3D,D]
    const float* Wo   = (const float*)d_in[2];   // [D,D]
    float* out = (float*)d_out;                  // [B,S,D]

    const int M = MB_B * MB_S;                   // 4096

    // workspace carve-up (~96 MB)
    char* ws = (char*)d_ws;
    unsigned short* Xbf    = (unsigned short*)ws;            // 8 MB
    unsigned short* Wqkvbf = Xbf + (size_t)M * MB_D;         // 6 MB
    unsigned short* Wobf   = Wqkvbf + (size_t)3 * MB_D * MB_D; // 2 MB
    float* qkv = (float*)(Wobf + (size_t)MB_D * MB_D);       // 48 MB
    unsigned short* Qb  = (unsigned short*)((char*)qkv + (size_t)M * 3 * MB_D * 4);
    unsigned short* Kb  = Qb + (size_t)M * MB_D;             // 8 MB each
    unsigned short* Vt  = Kb + (size_t)M * MB_D;
    unsigned short* Obf = Vt + (size_t)M * MB_D;

    // 1) convert X, Wqkv, Wo to bf16
    {
        int n4 = M * MB_D / 4;
        cvt_f32_bf16x4<<<(n4 + 255) / 256, 256, 0, stream>>>(X, Xbf, n4);
        n4 = 3 * MB_D * MB_D / 4;
        cvt_f32_bf16x4<<<(n4 + 255) / 256, 256, 0, stream>>>(Wqkv, Wqkvbf, n4);
        n4 = MB_D * MB_D / 4;
        cvt_f32_bf16x4<<<(n4 + 255) / 256, 256, 0, stream>>>(Wo, Wobf, n4);
    }
    // 2) qkv = X * Wqkv^T   (M=4096, N=3072, K=1024)
    {
        dim3 grid(3 * MB_D / BN, M / BM);
        gemm_bf16nt<<<grid, 256, 0, stream>>>(Xbf, Wqkvbf, qkv, M, 3 * MB_D, MB_D);
    }
    // 3) RoPE + pack Q,K,V(bf16, V transposed)
    {
        int total = MB_B * MB_H * MB_S * MB_DH;  // 4M threads
        rope_pack<<<total / 256, 256, 0, stream>>>(qkv, Qb, Kb, Vt);
    }
    // 4) causal flash attention -> Obf [B,S,D] bf16
    {
        int grid = MB_B * MB_H * (MB_S / 64);    // 1024 WGs x 128 threads
        flash_attn<<<grid, 128, 0, stream>>>(Qb, Kb, Vt, Obf);
    }
    // 5) out = Obf * Wo^T   (M=4096, N=1024, K=1024)
    {
        dim3 grid(MB_D / BN, M / BM);
        gemm_bf16nt<<<grid, 256, 0, stream>>>(Obf, Wobf, out, M, MB_D, MB_D);
    }
}
